// SemanticLevelContext_20109036880258
// MI455X (gfx1250) — compile-verified, matrix operand order checked by reading, not run on hardware
//
#include <hip/hip_runtime.h>

// ---------------- problem constants ----------------
#define BB    8
#define CC    512
#define HWP   4096      // H*W
#define KCLS  19
#define CTC   256
#define MTOT  (BB*HWP)  // 32768

typedef __attribute__((ext_vector_type(16))) _Float16 v16h;
typedef __attribute__((ext_vector_type(8)))  _Float16 h8;
typedef __attribute__((ext_vector_type(8)))  float    v8f;
typedef __attribute__((ext_vector_type(4)))  float    vf4;

// ---------------- small converters ----------------
__global__ void cvt_f32_to_f16(const float* __restrict__ src,
                               _Float16* __restrict__ dst, int n) {
  int i = blockIdx.x * 256 + threadIdx.x;
  if (i < n) dst[i] = (_Float16)src[i];
}

// x [B,C,HW] fp32 -> At [(b*HW+p)*C + c] f16  (32x32 tiled transpose)
__global__ void transpose_convert_x(const float* __restrict__ x,
                                    _Float16* __restrict__ At) {
  __shared__ float t[32][33];
  int b  = blockIdx.z;
  int p0 = blockIdx.x * 32;
  int c0 = blockIdx.y * 32;
  int tx = threadIdx.x, ty = threadIdx.y;   // block (32,8)
#pragma unroll
  for (int i = 0; i < 4; ++i) {
    int c = c0 + ty + i * 8;
    t[ty + i * 8][tx] = x[((size_t)b * CC + c) * HWP + p0 + tx];
  }
  __syncthreads();
#pragma unroll
  for (int i = 0; i < 4; ++i) {
    int p = p0 + ty + i * 8;
    At[((size_t)b * HWP + p) * CC + c0 + tx] = (_Float16)t[tx][ty + i * 8];
  }
}

// ---------------- stage A: argmax / class stats / pixel weights ----------------
__global__ void argmax_kernel(const float* __restrict__ preds,
                              int* __restrict__ cls, float* __restrict__ val) {
  int i = blockIdx.x * 256 + threadIdx.x;           // i = b*HW + p
  if (i >= MTOT) return;
  int b = i >> 12, p = i & (HWP - 1);
  const float* base = preds + (size_t)b * KCLS * HWP + p;
  float best = base[0];
  int   bk = 0;
#pragma unroll
  for (int k = 1; k < KCLS; ++k) {
    float v = base[(size_t)k * HWP];
    if (v > best) { best = v; bk = k; }               // first-max tie rule
  }
  cls[i] = bk; val[i] = best;
}

__global__ void class_stats(const int* __restrict__ cls, const float* __restrict__ val,
                            float* __restrict__ cM, float* __restrict__ cS,
                            float* __restrict__ cN) {
  __shared__ float red[256];
  int b = blockIdx.x / KCLS, k = blockIdx.x % KCLS;
  int t = threadIdx.x;
  const int base = b * HWP;
  float m = -3.4e38f;
  for (int p = t; p < HWP; p += 256)
    if (cls[base + p] == k) m = fmaxf(m, val[base + p]);
  red[t] = m; __syncthreads();
  for (int s = 128; s > 0; s >>= 1) { if (t < s) red[t] = fmaxf(red[t], red[t + s]); __syncthreads(); }
  float M = red[0]; __syncthreads();

  float sum = 0.f, cnt = 0.f;
  for (int p = t; p < HWP; p += 256)
    if (cls[base + p] == k) { sum += expf(val[base + p] - M); cnt += 1.f; }
  red[t] = sum; __syncthreads();
  for (int s = 128; s > 0; s >>= 1) { if (t < s) red[t] += red[t + s]; __syncthreads(); }
  float S = red[0]; __syncthreads();
  red[t] = cnt; __syncthreads();
  for (int s = 128; s > 0; s >>= 1) { if (t < s) red[t] += red[t + s]; __syncthreads(); }
  if (t == 0) {
    int i = b * KCLS + k;
    cM[i] = M; cS[i] = (S > 0.f) ? S : 1.f; cN[i] = red[0];
  }
}

__global__ void pixel_weight(const int* __restrict__ cls, const float* __restrict__ val,
                             const float* __restrict__ cM, const float* __restrict__ cS,
                             float* __restrict__ wgt) {
  int i = blockIdx.x * 256 + threadIdx.x;
  if (i >= MTOT) return;
  int b = i >> 12;
  int c = cls[i];
  wgt[i] = expf(val[i] - cM[b * KCLS + c]) / cS[b * KCLS + c];
}

// cls_feat[b,k,c] += sum_{p: cls=k} w[p] * x[b,c,p]
__global__ void cls_feat_accum(const float* __restrict__ x, const int* __restrict__ cls,
                               const float* __restrict__ wgt, float* __restrict__ cf) {
  __shared__ float acc[256 * KCLS];
  __shared__ float cw[128];
  __shared__ int   ccl[128];
  int t = threadIdx.x;
  int c = blockIdx.x * 256 + t;
  int b = blockIdx.z;
  int pbase = blockIdx.y * 1024;
  for (int i = t; i < 256 * KCLS; i += 256) acc[i] = 0.f;
  __syncthreads();
  for (int pc = 0; pc < 8; ++pc) {
    int p0 = pbase + pc * 128;
    if (t < 128) { cw[t] = wgt[b * HWP + p0 + t]; ccl[t] = cls[b * HWP + p0 + t]; }
    __syncthreads();
    const float* xrow = x + ((size_t)b * CC + c) * HWP + p0;
    for (int j = 0; j < 128; ++j)
      acc[t * KCLS + ccl[j]] += cw[j] * xrow[j];
    __syncthreads();
  }
#pragma unroll
  for (int k = 0; k < KCLS; ++k)
    atomicAdd(&cf[((size_t)b * KCLS + k) * CC + c], acc[t * KCLS + k]);
}

// ---------------- per-class key/value projections (tiny) ----------------
__global__ void kv_project(const float* __restrict__ cf,
                           const float* __restrict__ kw1, const float* __restrict__ kg1, const float* __restrict__ kb1,
                           const float* __restrict__ kw2, const float* __restrict__ kg2, const float* __restrict__ kb2,
                           const float* __restrict__ vw,  const float* __restrict__ vg,  const float* __restrict__ vb,
                           float* __restrict__ keyv, float* __restrict__ valv) {
  __shared__ float row[CC];
  __shared__ float t1[CTC];
  int bk = blockIdx.x, t = threadIdx.x;
  row[t]       = cf[(size_t)bk * CC + t];
  row[t + 256] = cf[(size_t)bk * CC + t + 256];
  __syncthreads();
  float d1 = 0.f, dv = 0.f;
  for (int c = 0; c < CC; ++c) {
    float r = row[c];
    d1 += kw1[(size_t)t * CC + c] * r;
    dv += vw [(size_t)t * CC + c] * r;
  }
  t1[t] = fmaxf(d1 * kg1[t] + kb1[t], 0.f);
  valv[(size_t)bk * CTC + t] = fmaxf(dv * vg[t] + vb[t], 0.f);
  __syncthreads();
  float d2 = 0.f;
  for (int c = 0; c < CTC; ++c) d2 += kw2[(size_t)t * CTC + c] * t1[c];
  keyv[(size_t)bk * CTC + t] = fmaxf(d2 * kg2[t] + kb2[t], 0.f);
}

// ---------------- WMMA GEMM: out = relu(g*(A[M,KD] x Wt[N,KD]^T) + b) ----------------
// block tile 128(M) x 128(N), 8 waves of 32x64, 2x4 fragments of v_wmma_f32_16x16x32_f16,
// register-staged software pipeline over K chunks of 32.
template <int KD, bool OUTH, bool TROUT>
__global__ __launch_bounds__(256) void gemm_proj(const _Float16* __restrict__ A,
                                                 const _Float16* __restrict__ Wt,
                                                 const float* __restrict__ gain,
                                                 const float* __restrict__ bias,
                                                 _Float16* __restrict__ outH,
                                                 float* __restrict__ outF, int N) {
  __shared__ __align__(16) _Float16 As[128][40];
  __shared__ __align__(16) _Float16 Bs[128][40];
  constexpr int NK = KD / 32;
  int t = threadIdx.x;
  int wave = t >> 5, lane = t & 31;
  int wm = wave & 3;            // 4 m-wave tiles of 32
  int wn = wave >> 2;           // 2 n-wave tiles of 64
  int lane16 = lane & 15;
  int laneHi = (lane >> 4) & 1;
  int c0 = laneHi ? 8 : 0;      // K sub-chunk per documented A/B lane layout
  int m0 = blockIdx.x * 128;
  int n0 = blockIdx.y * 128;

  // cooperative-load mapping: each thread owns 2 A chunks and 2 B chunks of 8 halves
  int lrow = t >> 2;            // 0..63
  int lcol = (t & 3) * 8;       // 0/8/16/24
  const _Float16* pA0 = A  + (size_t)(m0 + lrow)      * KD + lcol;
  const _Float16* pA1 = A  + (size_t)(m0 + 64 + lrow) * KD + lcol;
  const _Float16* pB0 = Wt + (size_t)(n0 + lrow)      * KD + lcol;
  const _Float16* pB1 = Wt + (size_t)(n0 + 64 + lrow) * KD + lcol;
  _Float16* sA0 = &As[lrow][lcol];
  _Float16* sA1 = &As[64 + lrow][lcol];
  _Float16* sB0 = &Bs[lrow][lcol];
  _Float16* sB1 = &Bs[64 + lrow][lcol];

  v8f acc[2][4] = {};
  // prologue: stage K-chunk 0 into registers
  vf4 rA0 = *(const vf4*)pA0;
  vf4 rA1 = *(const vf4*)pA1;
  vf4 rB0 = *(const vf4*)pB0;
  vf4 rB1 = *(const vf4*)pB1;

  for (int kc = 0; kc < NK; ++kc) {
    *(vf4*)sA0 = rA0; *(vf4*)sA1 = rA1;
    *(vf4*)sB0 = rB0; *(vf4*)sB1 = rB1;
    __syncthreads();
    // issue next chunk's global loads; wait is deferred to next iteration's ds_store
    if (kc + 1 < NK) {
      int ko = (kc + 1) * 32;
      rA0 = *(const vf4*)(pA0 + ko);
      rA1 = *(const vf4*)(pA1 + ko);
      rB0 = *(const vf4*)(pB0 + ko);
      rB1 = *(const vf4*)(pB1 + ko);
    }
    v16h aF[2], bF[4];
#pragma unroll
    for (int mf = 0; mf < 2; ++mf) {
      int row = wm * 32 + mf * 16 + lane16;
      h8 lo = *(const h8*)&As[row][c0];
      h8 hi = *(const h8*)&As[row][c0 + 16];
      aF[mf] = __builtin_shufflevector(lo, hi, 0,1,2,3,4,5,6,7,8,9,10,11,12,13,14,15);
    }
#pragma unroll
    for (int nf = 0; nf < 4; ++nf) {
      int row = wn * 64 + nf * 16 + lane16;
      h8 lo = *(const h8*)&Bs[row][c0];
      h8 hi = *(const h8*)&Bs[row][c0 + 16];
      bF[nf] = __builtin_shufflevector(lo, hi, 0,1,2,3,4,5,6,7,8,9,10,11,12,13,14,15);
    }
#pragma unroll
    for (int mf = 0; mf < 2; ++mf)
#pragma unroll
      for (int nf = 0; nf < 4; ++nf)
        acc[mf][nf] = __builtin_amdgcn_wmma_f32_16x16x32_f16(
            false, aF[mf], false, bF[nf], (short)0, acc[mf][nf], false, false);
    __syncthreads();
  }

  if constexpr (!TROUT) {
#pragma unroll
    for (int nf = 0; nf < 4; ++nf) {
      int n = n0 + wn * 64 + nf * 16 + lane16;
      float gg = gain[n], bb = bias[n];
#pragma unroll
      for (int mf = 0; mf < 2; ++mf) {
#pragma unroll
        for (int r = 0; r < 8; ++r) {
          int m = m0 + wm * 32 + mf * 16 + r + laneHi * 8;
          float v = fmaxf(acc[mf][nf][r] * gg + bb, 0.f);
          size_t idx = (size_t)m * N + n;
          if constexpr (OUTH) outH[idx] = (_Float16)v;
          else                outF[idx] = v;
        }
      }
    }
  } else {
    // transpose epilogue: out[b, n, p] coalesced along p
    __shared__ float Ts[128][129];
#pragma unroll
    for (int nf = 0; nf < 4; ++nf) {
      int nl = wn * 64 + nf * 16 + lane16;
      int n = n0 + nl;
      float gg = gain[n], bb = bias[n];
#pragma unroll
      for (int mf = 0; mf < 2; ++mf) {
#pragma unroll
        for (int r = 0; r < 8; ++r) {
          int ml = wm * 32 + mf * 16 + r + laneHi * 8;
          Ts[nl][ml] = fmaxf(acc[mf][nf][r] * gg + bb, 0.f);
        }
      }
    }
    __syncthreads();
    int b = m0 >> 12;         // M tiles never cross batch (4096 % 128 == 0)
    int p0 = m0 & (HWP - 1);
    for (int i = t; i < 128 * 128; i += 256) {
      int nl = i >> 7, j = i & 127;
      outF[((size_t)b * N + (n0 + nl)) * HWP + p0 + j] = Ts[nl][j];
    }
  }
}

// ---------------- collapsed attention ----------------
__global__ void attn_collapse(const _Float16* __restrict__ Qf,
                              const float* __restrict__ keyv,
                              const float* __restrict__ valv,
                              const float* __restrict__ cN,
                              _Float16* __restrict__ ctxH) {
  __shared__ float kk[KCLS * CTC];
  __shared__ float vv[KCLS * CTC];
  __shared__ float cn[KCLS];
  int t = threadIdx.x;
  int b = blockIdx.x >> 4;
  int p = ((blockIdx.x & 15) << 8) + t;
  for (int i = t; i < KCLS * CTC; i += 256) {
    kk[i] = keyv[(size_t)b * KCLS * CTC + i];
    vv[i] = valv[(size_t)b * KCLS * CTC + i];
  }
  if (t < KCLS) cn[t] = cN[b * KCLS + t];
  __syncthreads();

  const _Float16* q = Qf + ((size_t)(b * HWP) + p) * CTC;
  float s[KCLS];
#pragma unroll
  for (int k = 0; k < KCLS; ++k) s[k] = 0.f;
  for (int ct = 0; ct < CTC; ++ct) {
    float qv = (float)q[ct];
#pragma unroll
    for (int k = 0; k < KCLS; ++k) s[k] += qv * kk[k * CTC + ct];
  }
  float smax = -3.4e38f;
#pragma unroll
  for (int k = 0; k < KCLS; ++k)
    if (cn[k] > 0.f) smax = fmaxf(smax, s[k] * 0.0625f);
  float den = 0.f;
#pragma unroll
  for (int k = 0; k < KCLS; ++k) {
    float w = (cn[k] > 0.f) ? cn[k] * expf(s[k] * 0.0625f - smax) : 0.f;
    s[k] = w; den += w;
  }
  float inv = 1.f / den;
#pragma unroll
  for (int k = 0; k < KCLS; ++k) s[k] *= inv;

  _Float16* o = ctxH + ((size_t)(b * HWP) + p) * CTC;
  for (int ct = 0; ct < CTC; ++ct) {
    float a = 0.f;
#pragma unroll
    for (int k = 0; k < KCLS; ++k) a += s[k] * vv[k * CTC + ct];
    o[ct] = (_Float16)a;
  }
}

// ---------------- host launcher ----------------
extern "C" void kernel_launch(void* const* d_in, const int* in_sizes, int n_in,
                              void* d_out, int out_size, void* d_ws, size_t ws_size,
                              hipStream_t stream) {
  const float* x     = (const float*)d_in[0];
  const float* preds = (const float*)d_in[1];
  // d_in[2] = feats_il (unused by the reference)
  const float* q_w1 = (const float*)d_in[3];
  const float* q_g1 = (const float*)d_in[4];
  const float* q_b1 = (const float*)d_in[5];
  const float* q_w2 = (const float*)d_in[6];
  const float* q_g2 = (const float*)d_in[7];
  const float* q_b2 = (const float*)d_in[8];
  const float* k_w1 = (const float*)d_in[9];
  const float* k_g1 = (const float*)d_in[10];
  const float* k_b1 = (const float*)d_in[11];
  const float* k_w2 = (const float*)d_in[12];
  const float* k_g2 = (const float*)d_in[13];
  const float* k_b2 = (const float*)d_in[14];
  const float* v_w  = (const float*)d_in[15];
  const float* v_g  = (const float*)d_in[16];
  const float* v_b  = (const float*)d_in[17];
  const float* o_w  = (const float*)d_in[18];
  const float* o_g  = (const float*)d_in[19];
  const float* o_b  = (const float*)d_in[20];
  float* out = (float*)d_out;

  char* ws = (char*)d_ws;
  size_t off = 0;
  auto alloc = [&](size_t bytes) { size_t o = off; off += (bytes + 255) & ~size_t(255); return o; };
  size_t o_At  = alloc((size_t)MTOT * CC * 2);   // x^T f16 (aliased by ctx later)
  size_t o_Q1  = alloc((size_t)MTOT * CTC * 2);  // q1 activations f16
  size_t o_Qf  = alloc((size_t)MTOT * CTC * 2);  // q2 activations f16
  size_t o_w1h = alloc((size_t)CTC * CC * 2);
  size_t o_w2h = alloc((size_t)CTC * CTC * 2);
  size_t o_owh = alloc((size_t)CC * CTC * 2);
  size_t o_cls = alloc((size_t)MTOT * 4);
  size_t o_val = alloc((size_t)MTOT * 4);
  size_t o_wgt = alloc((size_t)MTOT * 4);
  size_t o_cM  = alloc(1024);
  size_t o_cS  = alloc(1024);
  size_t o_cN  = alloc(1024);
  size_t o_cf  = alloc((size_t)BB * KCLS * CC * 4);
  size_t o_key = alloc((size_t)BB * KCLS * CTC * 4);
  size_t o_vv  = alloc((size_t)BB * KCLS * CTC * 4);
  (void)ws_size; (void)in_sizes; (void)n_in; (void)out_size;

  _Float16* At   = (_Float16*)(ws + o_At);
  _Float16* ctxH = (_Float16*)(ws + o_At);   // alias: At dead after q1 GEMM
  _Float16* Q1h  = (_Float16*)(ws + o_Q1);
  _Float16* QfH  = (_Float16*)(ws + o_Qf);
  _Float16* w1h  = (_Float16*)(ws + o_w1h);
  _Float16* w2h  = (_Float16*)(ws + o_w2h);
  _Float16* owh  = (_Float16*)(ws + o_owh);
  int*   cls = (int*)(ws + o_cls);
  float* val = (float*)(ws + o_val);
  float* wgt = (float*)(ws + o_wgt);
  float* cM  = (float*)(ws + o_cM);
  float* cS  = (float*)(ws + o_cS);
  float* cN  = (float*)(ws + o_cN);
  float* cf  = (float*)(ws + o_cf);
  float* keyv = (float*)(ws + o_key);
  float* valv = (float*)(ws + o_vv);

  // weight conversions
  cvt_f32_to_f16<<<(CTC * CC + 255) / 256, 256, 0, stream>>>(q_w1, w1h, CTC * CC);
  cvt_f32_to_f16<<<(CTC * CTC + 255) / 256, 256, 0, stream>>>(q_w2, w2h, CTC * CTC);
  cvt_f32_to_f16<<<(CC * CTC + 255) / 256, 256, 0, stream>>>(o_w, owh, CC * CTC);

  // x transpose/convert
  transpose_convert_x<<<dim3(HWP / 32, CC / 32, BB), dim3(32, 8), 0, stream>>>(x, At);

  // class statistics path
  argmax_kernel<<<MTOT / 256, 256, 0, stream>>>(preds, cls, val);
  class_stats<<<BB * KCLS, 256, 0, stream>>>(cls, val, cM, cS, cN);
  pixel_weight<<<MTOT / 256, 256, 0, stream>>>(cls, val, cM, cS, wgt);
  hipMemsetAsync(cf, 0, (size_t)BB * KCLS * CC * 4, stream);
  cls_feat_accum<<<dim3(CC / 256, HWP / 1024, BB), 256, 0, stream>>>(x, cls, wgt, cf);

  // per-class key/value projections
  kv_project<<<BB * KCLS, 256, 0, stream>>>(cf, k_w1, k_g1, k_b1, k_w2, k_g2, k_b2,
                                            v_w, v_g, v_b, keyv, valv);

  // query projections (WMMA)
  gemm_proj<CC, true, false><<<dim3(MTOT / 128, CTC / 128), 256, 0, stream>>>(
      At, w1h, q_g1, q_b1, Q1h, nullptr, CTC);
  gemm_proj<CTC, true, false><<<dim3(MTOT / 128, CTC / 128), 256, 0, stream>>>(
      Q1h, w2h, q_g2, q_b2, QfH, nullptr, CTC);

  // collapsed attention -> ctx (f16, aliases At)
  attn_collapse<<<BB * 16, 256, 0, stream>>>(QfH, keyv, valv, cN, ctxH);

  // output projection (WMMA, transposed store to [B,C,H,W])
  gemm_proj<CTC, false, true><<<dim3(MTOT / 128, CC / 128), 256, 0, stream>>>(
      ctxH, owh, o_g, o_b, nullptr, out, CC);
}
